// MambaBlock_9182640079330
// MI455X (gfx1250) — matrix operand order repaired and checked by reading, not compiled, not run on hardware
//
#include <hip/hip_runtime.h>

// Mamba block forward for MI455X (gfx1250, wave32, WMMA), bf16 operand storage.
// Shapes fixed by the reference: B=2, L=1024, D_MODEL=1024, D_INNER=2048, N=16, DT_RANK=64.
// Pipeline: cvt(x,W*) -> bf16-WMMA GEMM (in-proj, bf16 out) -> conv+silu (bf16 out) ->
//           WMMA GEMM (x-proj, bf16 out, N-guarded) -> WMMA GEMM (dt-proj, softplus, f32 out)
//           -> selective scan (fp32 state, bf16 gated out) -> bf16-WMMA GEMM (out-proj, f32 out).
// GEMM staging uses CDNA5 async global->LDS loads (ASYNCcnt) when the toolchain
// exposes the builtins; otherwise falls back to reg-relay double buffering.

#define ML      2048   // B*L rows
#define DMODEL  1024
#define DINNER  2048
#define NST     16
#define DTRANK  64
#define XPN     96     // DT_RANK + 2*N

typedef __attribute__((ext_vector_type(16))) __bf16 v16bf;
typedef __attribute__((ext_vector_type(8)))  float  v8f;
typedef __attribute__((ext_vector_type(4)))  float  f32x4;
typedef __attribute__((ext_vector_type(4)))  unsigned int u32x4;
typedef __attribute__((ext_vector_type(2)))  unsigned int u32x2;
typedef unsigned short ushort_t;

#if __has_builtin(__builtin_amdgcn_global_load_async_to_lds_b128) && \
    __has_builtin(__builtin_amdgcn_s_wait_asynccnt)
#define HAVE_ASYNC_LDS 1
#else
#define HAVE_ASYNC_LDS 0
#endif

__device__ __forceinline__ unsigned short f2bf(float f) {
  unsigned u = __builtin_bit_cast(unsigned, f);
  u += 0x7FFFu + ((u >> 16) & 1u);          // round-to-nearest-even
  return (unsigned short)(u >> 16);
}
__device__ __forceinline__ unsigned pack2bf(float x, float y) {
  return (unsigned)f2bf(x) | ((unsigned)f2bf(y) << 16);
}
__device__ __forceinline__ float bf2f(unsigned short h) {
  return __builtin_bit_cast(float, (unsigned)h << 16);
}
__device__ __forceinline__ float silu_(float x)     { return x / (1.f + __expf(-x)); }
__device__ __forceinline__ float softplus_(float x) { return fmaxf(x, 0.f) + log1pf(__expf(-fabsf(x))); }

#if HAVE_ASYNC_LDS
// builtin signature (per hipcc diagnostic): (v4i AS1*, v4i AS3*, imm offset, imm cpol)
typedef int v4i_vs __attribute__((vector_size(16)));
typedef __attribute__((address_space(1))) v4i_vs* as1_v4i_ptr;
typedef __attribute__((address_space(3))) v4i_vs* as3_v4i_ptr;

__device__ __forceinline__ void async_b128(const ushort_t* g, ushort_t* l) {
  __builtin_amdgcn_global_load_async_to_lds_b128(
      (as1_v4i_ptr)(void*)g, (as3_v4i_ptr)(void*)l, 0, 0);
}
#endif

// ---------------------------------------------------------------------------
// fp32 -> bf16 bulk conversion (4 elements/thread, b128 in / b64 out)
// ---------------------------------------------------------------------------
__global__ __launch_bounds__(256)
void cvt_bf16_kernel(const float* __restrict__ in, ushort_t* __restrict__ out, int n4)
{
  int i = blockIdx.x * 256 + threadIdx.x;
  if (i < n4) {
    f32x4 v = *(const f32x4*)&in[(size_t)i * 4];
    u32x2 p = {pack2bf(v[0], v[1]), pack2bf(v[2], v[3])};
    *(u32x2*)&out[(size_t)i * 4] = p;
  }
}

// ---------------------------------------------------------------------------
// bf16-in, bf16-compute GEMM: C[M,N] = op(A[M,K] @ B[K,N]), fp32 accumulate.
// Block tile 64(M) x 128(N), K-step 32, double-buffered LDS (1 barrier/iter).
// 256 threads = 8 waves arranged 2(M) x 4(N); each wave = 2x2 WMMA 16x16x32 tiles.
// M multiple of 64, K multiple of 32 (true at all call sites); lda/ldb mult of 8.
// MODE 0: v            MODE 1: softplus(v + bias[col])
// OUTF 0: f32 store    OUTF 1: bf16 store
// NGUARD: bounds-check N (only for the x-proj GEMM, N=96).
// ---------------------------------------------------------------------------
#define LDA_S 40    // 64x32 A tile, padded row stride (bf16 units); 80B rows
#define LDB_S 136   // 32x128 B tile, padded; 272B rows

template <int MODE, int OUTF, bool NGUARD>
__global__ __launch_bounds__(256)
void gemm_bf16_wmma(const ushort_t* __restrict__ A, const ushort_t* __restrict__ B,
                    void* __restrict__ Cp, int M, int N, int K,
                    int lda, int ldb, int ldc, const float* __restrict__ bias)
{
  __shared__ __align__(16) ushort_t As[2][64 * LDA_S];
  __shared__ __align__(16) ushort_t Bs[2][32 * LDB_S];

  const int tid  = threadIdx.x;
  const int lane = tid & 31;
  const int lo   = lane & 15;
  const int hi   = lane >> 4;
  const int wid  = tid >> 5;
  const int wm   = (wid >> 2) << 5;   // 0,32
  const int wn   = (wid & 3)  << 5;   // 0,32,64,96

  const int m0 = blockIdx.y << 6;
  const int n0 = blockIdx.x << 7;

  // staging: A 64x32 = one b128 (8 bf16)/thread; B 32x128 = two b128/thread
  const int arow = tid >> 2;          // 0..63
  const int acol = (tid & 3) << 3;    // 0,8,16,24
  const int brow = tid >> 4;          // 0..15 (and +16)
  const int bcol = (tid & 15) << 3;   // 0,8,..,120
  const bool bok = (!NGUARD || (n0 + bcol) < N);

  const v8f vzero = {0.f,0.f,0.f,0.f,0.f,0.f,0.f,0.f};
  v8f acc[2][2];
#pragma unroll
  for (int i = 0; i < 2; ++i)
#pragma unroll
    for (int j = 0; j < 2; ++j) acc[i][j] = vzero;

  union ABfrag { v16bf v; u32x4 q[2]; };
  const u32x4 qzero = {0u, 0u, 0u, 0u};

#if HAVE_ASYNC_LDS
  // ---- prologue: async-stage tile 0 into buffer 0 (no VGPR round-trip) ----
  async_b128(&A[(size_t)(m0 + arow) * lda + acol], &As[0][arow * LDA_S + acol]);
  if (bok) {
    async_b128(&B[(size_t)brow * ldb + n0 + bcol],        &Bs[0][brow * LDB_S + bcol]);
    async_b128(&B[(size_t)(brow + 16) * ldb + n0 + bcol], &Bs[0][(brow + 16) * LDB_S + bcol]);
  } else {
    *(u32x4*)&Bs[0][brow * LDB_S + bcol]        = qzero;
    *(u32x4*)&Bs[0][(brow + 16) * LDB_S + bcol] = qzero;
  }
  __builtin_amdgcn_s_wait_asynccnt(0);
  __syncthreads();
#else
  u32x4 arv, brv[2];
  arv = *(const u32x4*)&A[(size_t)(m0 + arow) * lda + acol];
#pragma unroll
  for (int i = 0; i < 2; ++i)
    brv[i] = bok ? *(const u32x4*)&B[(size_t)(brow + 16 * i) * ldb + n0 + bcol] : qzero;
  *(u32x4*)&As[0][arow * LDA_S + acol] = arv;
#pragma unroll
  for (int i = 0; i < 2; ++i)
    *(u32x4*)&Bs[0][(brow + 16 * i) * LDB_S + bcol] = brv[i];
  __syncthreads();
#endif

  int cur = 0;
  for (int k0 = 0; k0 < K; k0 += 32) {
    const bool hasNext = (k0 + 32 < K);

    // ---- stage next tile into the other buffer (hidden behind WMMA work) ----
    if (hasNext) {
      const int nxt = cur ^ 1;
#if HAVE_ASYNC_LDS
      async_b128(&A[(size_t)(m0 + arow) * lda + (k0 + 32) + acol],
                 &As[nxt][arow * LDA_S + acol]);
      if (bok) {
        async_b128(&B[(size_t)(k0 + 32 + brow) * ldb + n0 + bcol],
                   &Bs[nxt][brow * LDB_S + bcol]);
        async_b128(&B[(size_t)(k0 + 32 + brow + 16) * ldb + n0 + bcol],
                   &Bs[nxt][(brow + 16) * LDB_S + bcol]);
      } else {
        *(u32x4*)&Bs[nxt][brow * LDB_S + bcol]        = qzero;
        *(u32x4*)&Bs[nxt][(brow + 16) * LDB_S + bcol] = qzero;
      }
#else
      arv = *(const u32x4*)&A[(size_t)(m0 + arow) * lda + (k0 + 32) + acol];
#pragma unroll
      for (int i = 0; i < 2; ++i)
        brv[i] = bok ? *(const u32x4*)&B[(size_t)(k0 + 32 + brow + 16 * i) * ldb + n0 + bcol]
                     : qzero;
      (void)nxt;
#endif
      if (k0 + 64 < K) {  // L2 warm-up two tiles ahead (global_prefetch_b8)
        __builtin_prefetch(&A[(size_t)(m0 + arow) * lda + (k0 + 64) + acol], 0, 1);
        __builtin_prefetch(&B[(size_t)(k0 + 64 + brow) * ldb + n0 + (NGUARD ? 0 : bcol)], 0, 1);
      }
    }

    // ---- fragments per ISA §7.12.2 ----
    // A (16x32 bf16): lane -> row M=lo; VGPR0..3 = K 8*hi.., VGPR4..7 = K 16+8*hi..
    ABfrag a[2], b[2];
#pragma unroll
    for (int tm = 0; tm < 2; ++tm) {
      const ushort_t* ap = &As[cur][(wm + (tm << 4) + lo) * LDA_S];
      a[tm].q[0] = *(const u32x4*)(ap + (hi << 3));
      a[tm].q[1] = *(const u32x4*)(ap + 16 + (hi << 3));
    }
    // B (32x16 bf16): lane -> col N=lo; elems 0..7 = K 8*hi.., 8..15 = K 16+8*hi..
#pragma unroll
    for (int tn = 0; tn < 2; ++tn) {
      const ushort_t* bp = &Bs[cur][wn + (tn << 4) + lo];
#pragma unroll
      for (int s = 0; s < 16; ++s) {
        const int kk = (s < 8) ? ((hi << 3) + s) : (16 + (hi << 3) + (s - 8));
        b[tn].v[s] = __builtin_bit_cast(__bf16, bp[kk * LDB_S]);
      }
    }
#pragma unroll
    for (int tm = 0; tm < 2; ++tm)
#pragma unroll
      for (int tn = 0; tn < 2; ++tn)
        acc[tm][tn] = __builtin_amdgcn_wmma_f32_16x16x32_bf16(
            false, a[tm].v, false, b[tn].v, (short)0, acc[tm][tn], false, false);

    // ---- publish next buffer ----
    if (hasNext) {
      const int nxt = cur ^ 1;
#if HAVE_ASYNC_LDS
      __builtin_amdgcn_s_wait_asynccnt(0);   // async LDS writes landed
#else
      *(u32x4*)&As[nxt][arow * LDA_S + acol] = arv;
#pragma unroll
      for (int i = 0; i < 2; ++i)
        *(u32x4*)&Bs[nxt][(brow + 16 * i) * LDB_S + bcol] = brv[i];
#endif
      __syncthreads();
      cur = nxt;
    }
  }

  // ---- epilogue: C layout M = r + 8*hi, N = lo ----
  float*    Cf = (float*)Cp;
  ushort_t* Cb = (ushort_t*)Cp;
#pragma unroll
  for (int tm = 0; tm < 2; ++tm)
#pragma unroll
    for (int tn = 0; tn < 2; ++tn) {
      const int gcol = n0 + wn + (tn << 4) + lo;
      if (NGUARD && gcol >= N) continue;
      const float bv = (MODE == 1) ? bias[gcol] : 0.f;
#pragma unroll
      for (int r = 0; r < 8; ++r) {
        const int grow = m0 + wm + (tm << 4) + r + (hi << 3);
        float v = acc[tm][tn][r];
        if (MODE == 1) v = softplus_(v + bv);
        if (OUTF == 0) Cf[(size_t)grow * ldc + gcol] = v;
        else           Cb[(size_t)grow * ldc + gcol] = f2bf(v);
      }
    }
}

// ---------------------------------------------------------------------------
// Causal depthwise conv (k=4) + SiLU; 2 channels/thread (paired bf16 I/O).
//   u = silu(conv(xp)+b_conv), sres = silu(res);  xr is bf16 [ML, 2*DINNER].
// ---------------------------------------------------------------------------
__global__ __launch_bounds__(256)
void conv_silu_kernel(const ushort_t* __restrict__ xr, const float* __restrict__ Wc,
                      const float* __restrict__ bc,
                      ushort_t* __restrict__ u_bf, ushort_t* __restrict__ sres_bf)
{
  int idx = blockIdx.x * 256 + threadIdx.x;   // ML * DINNER/2 threads
  int dp  = idx & (DINNER / 2 - 1);
  int m   = idx >> 10;
  int b = m >> 10, l = m & 1023;
  int d = dp << 1;

  float a0 = bc[d], a1 = bc[d + 1];
#pragma unroll
  for (int t = 0; t < 4; ++t) {
    int ls = l - 3 + t;
    if (ls >= 0) {
      unsigned pr = *(const unsigned*)&xr[(size_t)((b << 10) + ls) * (2 * DINNER) + d];
      a0 += bf2f((unsigned short)pr) * Wc[t * DINNER + d];
      a1 += bf2f((unsigned short)(pr >> 16)) * Wc[t * DINNER + d + 1];
    }
  }
  *(unsigned*)&u_bf[(size_t)m * DINNER + d] = pack2bf(silu_(a0), silu_(a1));

  unsigned rr = *(const unsigned*)&xr[(size_t)m * (2 * DINNER) + DINNER + d];
  *(unsigned*)&sres_bf[(size_t)m * DINNER + d] =
      pack2bf(silu_(bf2f((unsigned short)rr)), silu_(bf2f((unsigned short)(rr >> 16))));
}

// ---------------------------------------------------------------------------
// Selective scan. One block = 256 channels of one batch; fp32 state (16 per
// thread) + 16 A coefficients in VGPRs, sweeps L sequentially. B/C vectors
// staged (bf16 -> f32) in LDS in chunks of 16 timesteps. Fused output gate:
//   yg = (y_scan + u*D) * silu(res), emitted in bf16 for the out-projection.
// ---------------------------------------------------------------------------
#define SCHUNK 16

__global__ __launch_bounds__(256)
void scan_kernel(const float* __restrict__ delta, const ushort_t* __restrict__ u_bf,
                 const ushort_t* __restrict__ sres_bf, const ushort_t* __restrict__ xdbl_bf,
                 const float* __restrict__ A_log, const float* __restrict__ Dp,
                 ushort_t* __restrict__ yg_bf)
{
  __shared__ float BC[SCHUNK][32];    // per step: [0..15]=B, [16..31]=C
  const int b = blockIdx.x >> 3;
  const int d = ((blockIdx.x & 7) << 8) + threadIdx.x;

  float Aa[NST], s[NST];
#pragma unroll
  for (int n = 0; n < NST; ++n) { Aa[n] = -__expf(A_log[d * NST + n]); s[n] = 0.f; }
  const float Dv = Dp[d];

  for (int l0 = 0; l0 < 1024; l0 += SCHUNK) {
#pragma unroll
    for (int i = 0; i < 2; ++i) {     // 512 values, 256 threads
      int e = threadIdx.x + (i << 8);
      int o = e >> 5, j = e & 31;
      BC[o][j] = bf2f(xdbl_bf[(size_t)((b << 10) + l0 + o) * XPN + DTRANK + j]);
    }
    __syncthreads();

    for (int o = 0; o < SCHUNK; ++o) {
      const int m = (b << 10) + l0 + o;
      float dl = delta[(size_t)m * DINNER + d];
      float uu = bf2f(u_bf[(size_t)m * DINNER + d]);
      float du = dl * uu;
      float y = 0.f;
#pragma unroll
      for (int n = 0; n < NST; ++n) {
        float dA = __expf(dl * Aa[n]);
        s[n] = dA * s[n] + du * BC[o][n];
        y += s[n] * BC[o][16 + n];
      }
      float g = (y + uu * Dv) * bf2f(sres_bf[(size_t)m * DINNER + d]);
      yg_bf[(size_t)m * DINNER + d] = f2bf(g);
    }
    __syncthreads();
  }
}

// ---------------------------------------------------------------------------
extern "C" void kernel_launch(void* const* d_in, const int* in_sizes, int n_in,
                              void* d_out, int out_size, void* d_ws, size_t ws_size,
                              hipStream_t stream) {
  (void)in_sizes; (void)n_in; (void)out_size; (void)ws_size;
  const float* x      = (const float*)d_in[0];
  const float* W_in   = (const float*)d_in[1];
  const float* W_conv = (const float*)d_in[2];
  const float* b_conv = (const float*)d_in[3];
  const float* W_xproj= (const float*)d_in[4];
  const float* W_dt   = (const float*)d_in[5];
  const float* b_dt   = (const float*)d_in[6];
  const float* A_log  = (const float*)d_in[7];
  const float* Dp     = (const float*)d_in[8];
  const float* W_out  = (const float*)d_in[9];
  float* out = (float*)d_out;

  // ---- workspace: fp32 delta first, then bf16 region (all 16B aligned) ----
  float*    delta  = (float*)d_ws;                       // 2048 x 2048 f32
  ushort_t* bw     = (ushort_t*)(delta + (size_t)ML * DINNER);
  ushort_t* x_bf   = bw;                                 // 2048 x 1024
  ushort_t* Win_bf = x_bf   + (size_t)ML * DMODEL;       // 1024 x 4096
  ushort_t* Wxp_bf = Win_bf + (size_t)DMODEL * 2 * DINNER; // 2048 x 96
  ushort_t* Wdt_bf = Wxp_bf + (size_t)DINNER * XPN;      // 64 x 2048
  ushort_t* Wout_bf= Wdt_bf + (size_t)DTRANK * DINNER;   // 2048 x 1024
  ushort_t* xr_bf  = Wout_bf+ (size_t)DINNER * DMODEL;   // 2048 x 4096
  ushort_t* u_bf   = xr_bf  + (size_t)ML * 2 * DINNER;   // 2048 x 2048
  ushort_t* sres_bf= u_bf   + (size_t)ML * DINNER;       // 2048 x 2048
  ushort_t* xdbl_bf= sres_bf+ (size_t)ML * DINNER;       // 2048 x 96
  ushort_t* yg_bf  = xdbl_bf+ (size_t)ML * XPN;          // 2048 x 2048

  dim3 blk(256);
  auto cvt = [&](const float* src, ushort_t* dst, size_t n) {
    int n4 = (int)(n / 4);
    cvt_bf16_kernel<<<(n4 + 255) / 256, blk, 0, stream>>>(src, dst, n4);
  };

  // 0) one-time fp32 -> bf16 conversions
  cvt(x,       x_bf,    (size_t)ML * DMODEL);
  cvt(W_in,    Win_bf,  (size_t)DMODEL * 2 * DINNER);
  cvt(W_xproj, Wxp_bf,  (size_t)DINNER * XPN);
  cvt(W_dt,    Wdt_bf,  (size_t)DTRANK * DINNER);
  cvt(W_out,   Wout_bf, (size_t)DINNER * DMODEL);

  // 1) xr = x @ W_in                       (M=2048, N=4096, K=1024) -> bf16
  gemm_bf16_wmma<0, 1, false><<<dim3(2 * DINNER / 128, ML / 64), blk, 0, stream>>>(
      x_bf, Win_bf, xr_bf, ML, 2 * DINNER, DMODEL, DMODEL, 2 * DINNER, 2 * DINNER, nullptr);

  // 2) u = silu(conv(xp)+b),  sres = silu(res)   -> bf16
  conv_silu_kernel<<<(ML * DINNER / 2) / 256, blk, 0, stream>>>(
      xr_bf, W_conv, b_conv, u_bf, sres_bf);

  // 3) xdbl = u @ W_xproj                  (M=2048, N=96, K=2048) -> bf16
  gemm_bf16_wmma<0, 1, true><<<dim3(1, ML / 64), blk, 0, stream>>>(
      u_bf, Wxp_bf, xdbl_bf, ML, XPN, DINNER, DINNER, XPN, XPN, nullptr);

  // 4) delta = softplus(xdbl[:,:64] @ W_dt + b_dt)  (M=2048, N=2048, K=64) -> f32
  gemm_bf16_wmma<1, 0, false><<<dim3(DINNER / 128, ML / 64), blk, 0, stream>>>(
      xdbl_bf, Wdt_bf, delta, ML, DINNER, DTRANK, XPN, DINNER, DINNER, b_dt);

  // 5) selective scan + gating -> yg (bf16)
  scan_kernel<<<2 * (DINNER / 256), blk, 0, stream>>>(
      delta, u_bf, sres_bf, xdbl_bf, A_log, Dp, yg_bf);

  // 6) out = yg @ W_out                    (M=2048, N=1024, K=2048) -> f32
  gemm_bf16_wmma<0, 0, false><<<dim3(DMODEL / 128, ML / 64), blk, 0, stream>>>(
      yg_bf, Wout_bf, out, ML, DMODEL, DINNER, DINNER, DMODEL, DMODEL, nullptr);
}
